// Attention_52175262712194
// MI455X (gfx1250) — compile-verified, matrix operand order checked by reading, not run on hardware
//
#include <hip/hip_runtime.h>
#include <hip/hip_bf16.h>
#include <math.h>

// ---------------- problem constants ----------------
#define BATCH 1024
#define TSEQ  64
#define FEAT  2048
#define GRU   1024
#define ATT   512

#define LDS_STRIDE (FEAT + 8)          // padded bf16 row stride (bank-conflict avoidance)
#define SMEM_BYTES (TSEQ * LDS_STRIDE * 2 + TSEQ * 4 * 2)  // hs_bf16 + logits + weights

typedef __bf16 v4bf  __attribute__((ext_vector_type(4)));
typedef __bf16 v8bf  __attribute__((ext_vector_type(8)));
typedef __bf16 v16bf __attribute__((ext_vector_type(16)));
typedef float  v8f   __attribute__((ext_vector_type(8)));

__device__ __forceinline__ v16bf cat16(v8bf lo, v8bf hi) {
    return __builtin_shufflevector(lo, hi, 0,1,2,3,4,5,6,7,8,9,10,11,12,13,14,15);
}

// ---------------- prep: dtype conversions / transposes ----------------
__global__ void cvt_query_kernel(const float* __restrict__ q, __bf16* __restrict__ qb) {
    int i = blockIdx.x * blockDim.x + threadIdx.x;          // GRU*BATCH elements
    qb[i] = (__bf16)q[i];
}

// W1 [G, A] -> W1T bf16 [A, G]
__global__ void cvt_w1t_kernel(const float* __restrict__ w1, __bf16* __restrict__ w1t) {
    int i = blockIdx.x * blockDim.x + threadIdx.x;          // ATT*GRU elements
    int n = i >> 10;              // /GRU
    int k = i & (GRU - 1);
    w1t[i] = (__bf16)w1[k * ATT + n];
}

// W2 [F, A] -> W2T bf16 [A, F]
__global__ void cvt_w2t_kernel(const float* __restrict__ w2, __bf16* __restrict__ w2t) {
    int i = blockIdx.x * blockDim.x + threadIdx.x;          // ATT*FEAT elements
    int n = i >> 11;              // /FEAT
    int k = i & (FEAT - 1);
    w2t[i] = (__bf16)w2[k * ATT + n];
}

// ---------------- prep: w1full = query @ W1 + b1 + b2  (bf16 WMMA) ----------------
// output [BATCH, ATT] f32.  Tiles: 64 x 32 of 16x16; 8 waves/WG, 1 tile/wave, 256 WGs.
__global__ __launch_bounds__(256) void w1_gemm_kernel(
        const __bf16* __restrict__ qb,   // [BATCH, GRU]
        const __bf16* __restrict__ w1t,  // [ATT, GRU]
        const float*  __restrict__ b1,
        const float*  __restrict__ b2,
        float* __restrict__ w1full)      // [BATCH, ATT]
{
    const int tid  = threadIdx.x;
    const int lane = tid & 31;
    const int lr   = lane & 15;
    const int h    = lane >> 4;
    const int tile = blockIdx.x * 8 + (tid >> 5);   // 0..2047
    const int mt   = tile >> 5;                      // 0..63
    const int nt   = tile & 31;                      // 0..31

    const __bf16* arow = qb  + (size_t)(mt * 16 + lr) * GRU;
    const __bf16* brow = w1t + (size_t)(nt * 16 + lr) * GRU;

    v8f acc = {};
    #pragma unroll 4
    for (int kk = 0; kk < GRU; kk += 32) {
        const int kb = kk + h * 8;
        v16bf a = cat16(*(const v8bf*)(arow + kb), *(const v8bf*)(arow + kb + 16));
        v16bf b = cat16(*(const v8bf*)(brow + kb), *(const v8bf*)(brow + kb + 16));
        acc = __builtin_amdgcn_wmma_f32_16x16x32_bf16(false, a, false, b,
                                                      (short)0, acc, false, false);
    }

    const int n = nt * 16 + lr;
    const float bias = b1[n] + b2[n];
    #pragma unroll
    for (int v = 0; v < 8; ++v) {
        const int m = mt * 16 + h * 8 + v;           // C/D layout: lanes16-31 hold M=8..15
        w1full[(size_t)m * ATT + n] = acc[v] + bias;
    }
}

// ---------------- main fused attention kernel ----------------
// 1 WG (512 thr / 16 waves) per batch element.
// Wave tiling: each wave owns ALL 4 M-tiles (T=64) x 2 N-tiles, so the 16 waves
// read each W2T element from L2 exactly once per workgroup (A reuse lives in LDS).
__global__ __launch_bounds__(512) void attn_fused_kernel(
        const float* __restrict__ hs,      // [B, T, F] f32
        const __bf16* __restrict__ w2t,    // [ATT, FEAT] bf16
        const float* __restrict__ w1full,  // [B, ATT] f32 (biases folded)
        const float* __restrict__ Vw,      // [ATT] f32
        float* __restrict__ outCtx,        // [B, F]
        float* __restrict__ outAttn)       // [B, T]
{
    extern __shared__ char smem[];
    __bf16* sH     = (__bf16*)smem;                               // [T][LDS_STRIDE]
    float*  sLogit = (float*)(smem + TSEQ * LDS_STRIDE * 2);      // [T]
    float*  sW     = sLogit + TSEQ;                               // [T]

    const int b   = blockIdx.x;
    const int tid = threadIdx.x;

    if (tid < TSEQ) sLogit[tid] = 0.0f;

    // ---- phase 1: stream hs[b] (512 KB) once, convert f32 -> bf16 into LDS ----
    {
        const float* src = hs + (size_t)b * TSEQ * FEAT + tid * 4;
        #pragma unroll 4
        for (int t = 0; t < TSEQ; ++t) {
            float4 hv = *(const float4*)(src + (size_t)t * FEAT);
            v4bf pk = { (__bf16)hv.x, (__bf16)hv.y, (__bf16)hv.z, (__bf16)hv.w };
            *(v4bf*)(sH + t * LDS_STRIDE + tid * 4) = pk;
        }
    }
    __syncthreads();

    // ---- phase 2: 64x512x2048 bf16 WMMA GEMM ----
    const int wv   = tid >> 5;           // wave id 0..15
    const int lane = tid & 31;
    const int lr   = lane & 15;
    const int h    = lane >> 4;

    const __bf16* arow0 = sH + (size_t)(0 * 16 + lr) * LDS_STRIDE;
    const __bf16* arow1 = sH + (size_t)(1 * 16 + lr) * LDS_STRIDE;
    const __bf16* arow2 = sH + (size_t)(2 * 16 + lr) * LDS_STRIDE;
    const __bf16* arow3 = sH + (size_t)(3 * 16 + lr) * LDS_STRIDE;
    const __bf16* brow0 = w2t + (size_t)((wv * 2 + 0) * 16 + lr) * FEAT;
    const __bf16* brow1 = w2t + (size_t)((wv * 2 + 1) * 16 + lr) * FEAT;

    v8f acc[2][4] = {};
    for (int kk = 0; kk < FEAT; kk += 32) {
        const int kb = kk + h * 8;
        // A-fragments: lanes0-15 hold K{kb..kb+7, kb+16..kb+23}; lanes16-31 shifted by 8.
        v16bf a0 = cat16(*(const v8bf*)(arow0 + kb), *(const v8bf*)(arow0 + kb + 16));
        v16bf a1 = cat16(*(const v8bf*)(arow1 + kb), *(const v8bf*)(arow1 + kb + 16));
        v16bf a2 = cat16(*(const v8bf*)(arow2 + kb), *(const v8bf*)(arow2 + kb + 16));
        v16bf a3 = cat16(*(const v8bf*)(arow3 + kb), *(const v8bf*)(arow3 + kb + 16));
        v16bf b0 = cat16(*(const v8bf*)(brow0 + kb), *(const v8bf*)(brow0 + kb + 16));
        v16bf b1 = cat16(*(const v8bf*)(brow1 + kb), *(const v8bf*)(brow1 + kb + 16));

        acc[0][0] = __builtin_amdgcn_wmma_f32_16x16x32_bf16(false, a0, false, b0, (short)0, acc[0][0], false, false);
        acc[0][1] = __builtin_amdgcn_wmma_f32_16x16x32_bf16(false, a1, false, b0, (short)0, acc[0][1], false, false);
        acc[0][2] = __builtin_amdgcn_wmma_f32_16x16x32_bf16(false, a2, false, b0, (short)0, acc[0][2], false, false);
        acc[0][3] = __builtin_amdgcn_wmma_f32_16x16x32_bf16(false, a3, false, b0, (short)0, acc[0][3], false, false);
        acc[1][0] = __builtin_amdgcn_wmma_f32_16x16x32_bf16(false, a0, false, b1, (short)0, acc[1][0], false, false);
        acc[1][1] = __builtin_amdgcn_wmma_f32_16x16x32_bf16(false, a1, false, b1, (short)0, acc[1][1], false, false);
        acc[1][2] = __builtin_amdgcn_wmma_f32_16x16x32_bf16(false, a2, false, b1, (short)0, acc[1][2], false, false);
        acc[1][3] = __builtin_amdgcn_wmma_f32_16x16x32_bf16(false, a3, false, b1, (short)0, acc[1][3], false, false);
    }

    // ---- phase 3: score = tanh(w2 + w1full); logits += score . V ----
    {
        float row[4][8] = {};
        #pragma unroll
        for (int j = 0; j < 2; ++j) {
            const int n = (wv * 2 + j) * 16 + lr;
            const float w1v = w1full[(size_t)b * ATT + n];
            const float vv  = Vw[n];
            #pragma unroll
            for (int mi = 0; mi < 4; ++mi)
                #pragma unroll
                for (int v = 0; v < 8; ++v)
                    row[mi][v] += tanhf(acc[j][mi][v] + w1v) * vv;
        }
        // reduce over the 16 N-lanes of each half-wave
        #pragma unroll
        for (int off = 1; off < 16; off <<= 1)
            #pragma unroll
            for (int mi = 0; mi < 4; ++mi)
                #pragma unroll
                for (int v = 0; v < 8; ++v)
                    row[mi][v] += __shfl_xor(row[mi][v], off, 32);
        if (lr == 0) {
            #pragma unroll
            for (int mi = 0; mi < 4; ++mi)
                #pragma unroll
                for (int v = 0; v < 8; ++v)
                    atomicAdd(&sLogit[mi * 16 + h * 8 + v], row[mi][v]);  // ds_add_f32
        }
    }
    __syncthreads();

    // ---- phase 4: softmax over T=64 (one wave; bV shifts all logits -> cancels) ----
    if (tid < 32) {
        float l0 = sLogit[tid], l1 = sLogit[tid + 32];
        float mx = fmaxf(l0, l1);
        #pragma unroll
        for (int off = 1; off < 32; off <<= 1) mx = fmaxf(mx, __shfl_xor(mx, off, 32));
        float e0 = __expf(l0 - mx), e1 = __expf(l1 - mx);
        float s = e0 + e1;
        #pragma unroll
        for (int off = 1; off < 32; off <<= 1) s += __shfl_xor(s, off, 32);
        const float inv = 1.0f / s;
        sW[tid]      = e0 * inv;
        sW[tid + 32] = e1 * inv;
        outAttn[(size_t)b * TSEQ + tid]      = e0 * inv;
        outAttn[(size_t)b * TSEQ + tid + 32] = e1 * inv;
    }
    __syncthreads();

    // ---- phase 5: context = sum_t w[t] * hs[b,t,:]  (from LDS, no HBM re-read) ----
    {
        const __bf16* col = sH + tid * 4;
        float4 ctx = {0.f, 0.f, 0.f, 0.f};
        #pragma unroll 4
        for (int t = 0; t < TSEQ; ++t) {
            const float wt = sW[t];
            v4bf hv = *(const v4bf*)(col + (size_t)t * LDS_STRIDE);
            ctx.x += wt * (float)hv[0];
            ctx.y += wt * (float)hv[1];
            ctx.z += wt * (float)hv[2];
            ctx.w += wt * (float)hv[3];
        }
        *(float4*)(outCtx + (size_t)b * FEAT + tid * 4) = ctx;
    }
}

// ---------------- host launcher ----------------
extern "C" void kernel_launch(void* const* d_in, const int* in_sizes, int n_in,
                              void* d_out, int out_size, void* d_ws, size_t ws_size,
                              hipStream_t stream) {
    const float* query = (const float*)d_in[0];   // [B, G]
    const float* hs    = (const float*)d_in[1];   // [B, T, F]
    const float* W1    = (const float*)d_in[2];   // [G, A]
    const float* b1    = (const float*)d_in[3];   // [A]
    const float* W2    = (const float*)d_in[4];   // [F, A]
    const float* b2    = (const float*)d_in[5];   // [A]
    const float* Vw    = (const float*)d_in[6];   // [A, 1]
    // d_in[7] = bV: shifts all logits equally over T -> cancels in softmax.

    float* outCtx  = (float*)d_out;                       // [B, F]
    float* outAttn = outCtx + (size_t)BATCH * FEAT;       // [B, T]

    // workspace layout
    char* ws = (char*)d_ws;
    __bf16* w2t    = (__bf16*)ws;                                  // ATT*FEAT bf16 (2 MB)
    __bf16* qb     = (__bf16*)(ws + (size_t)ATT * FEAT * 2);       // B*G bf16    (2 MB)
    __bf16* w1t    = (__bf16*)(ws + (size_t)ATT * FEAT * 2
                                  + (size_t)BATCH * GRU * 2);      // ATT*GRU bf16(1 MB)
    float*  w1full = (float*)(ws + (size_t)ATT * FEAT * 2
                                 + (size_t)BATCH * GRU * 2
                                 + (size_t)ATT * GRU * 2);         // B*ATT f32   (2 MB)

    (void)in_sizes; (void)n_in; (void)out_size; (void)ws_size;

    // allow >64KB dynamic LDS for the fused kernel (CDNA5: up to 320KB/WG)
    (void)hipFuncSetAttribute((const void*)attn_fused_kernel,
                              hipFuncAttributeMaxDynamicSharedMemorySize, SMEM_BYTES);

    cvt_query_kernel<<<(BATCH * GRU) / 256, 256, 0, stream>>>(query, qb);
    cvt_w1t_kernel  <<<(ATT * GRU)   / 256, 256, 0, stream>>>(W1, w1t);
    cvt_w2t_kernel  <<<(ATT * FEAT)  / 256, 256, 0, stream>>>(W2, w2t);

    w1_gemm_kernel<<<256, 256, 0, stream>>>(qb, w1t, b1, b2, w1full);

    attn_fused_kernel<<<BATCH, 512, SMEM_BYTES, stream>>>(hs, w2t, w1full, Vw,
                                                          outCtx, outAttn);
}
